// AKGCN_41609643164193
// MI455X (gfx1250) — matrix operand order repaired and checked by reading, not compiled, not run on hardware
//
#include <hip/hip_runtime.h>
#include <hip/hip_bf16.h>
#include <math.h>

// Problem constants (match reference)
#define BB 32
#define TT 16
#define NN 512
#define CC 128
#define KK 2

typedef _Float16 half16 __attribute__((ext_vector_type(16)));
typedef _Float16 half8  __attribute__((ext_vector_type(8)));
typedef float    float8 __attribute__((ext_vector_type(8)));

__device__ __forceinline__ half16 h16_combine(half8 lo, half8 hi) {
    return __builtin_shufflevector(lo, hi, 0,1,2,3,4,5,6,7,8,9,10,11,12,13,14,15);
}
__device__ __forceinline__ half8 load_h8(const _Float16* p) {
    return *(const half8*)p;
}
__device__ __forceinline__ half8 cvt_f32_to_h8(const float* p) {
    float8 f = *(const float8*)p;
    half8 h;
    #pragma unroll
    for (int i = 0; i < 8; ++i) h[i] = (_Float16)f[i];
    return h;
}

// ---------------------------------------------------------------------------
// theta_T[k][d][c] = theta[k][c][d]  (f32 -> f16, tiny)
// grid: (CC, KK), block: CC
// ---------------------------------------------------------------------------
__global__ void akgcn_theta_prep(const float* __restrict__ theta,
                                 _Float16* __restrict__ thT) {
    const int d = blockIdx.x;
    const int k = blockIdx.y;
    const int c = threadIdx.x;
    thT[((size_t)k * CC + d) * CC + c] =
        (_Float16)theta[(size_t)k * CC * CC + (size_t)c * CC + d];
}

// ---------------------------------------------------------------------------
// V[b][n][m] = row-normalized( ((2L-2)/L) * I + (2/L) * adj ), f16 output.
// grid: BB*NN blocks, block 256 (each thread: 2 of 512 row elements)
// ---------------------------------------------------------------------------
__global__ __launch_bounds__(256)
void akgcn_compute_v(const float* __restrict__ adj,
                     const float* __restrict__ lambdas,
                     int k,
                     _Float16* __restrict__ V) {
    const int row = blockIdx.x;
    const int b = row >> 9;
    const int n = row & (NN - 1);
    const float* arow = adj + ((size_t)b * NN + n) * NN;

    const float lam_raw = lambdas[k];
    const float lam = 1.0f + fmaxf(lam_raw, 0.0f);
    const float ce = (2.0f * lam - 2.0f) / lam;
    const float ca = 2.0f / lam;

    const int tid = threadIdx.x;
    const int m0 = tid;
    const int m1 = tid + 256;
    float w0 = ca * arow[m0] + ((m0 == n) ? ce : 0.0f);
    float w1 = ca * arow[m1] + ((m1 == n) ? ce : 0.0f);

    float s = w0 + w1;
    #pragma unroll
    for (int off = 16; off > 0; off >>= 1) s += __shfl_down(s, off, 32);
    __shared__ float parts[8];
    const int wv = tid >> 5, ln = tid & 31;
    if (ln == 0) parts[wv] = s;
    __syncthreads();
    if (tid == 0) {
        float t = 0.0f;
        #pragma unroll
        for (int i = 0; i < 8; ++i) t += parts[i];
        parts[0] = t;
    }
    __syncthreads();
    const float inv = 1.0f / parts[0];

    _Float16* vrow = V + ((size_t)b * NN + n) * NN;
    vrow[m0] = (_Float16)(w0 * inv);
    vrow[m1] = (_Float16)(w1 * inv);
}

// ---------------------------------------------------------------------------
// GEMM1: h[m,d] = sum_c A[m,c] * theta[c,d]   (A = x f32 or hid f16)
// theta_T staged fully in LDS (shared by all 8 waves). A fragment prefetched
// one k-step ahead from global. All 8 B tiles of a k-step are loaded into
// distinct registers BEFORE the WMMA group so the ds waits stagger.
// Output stored transposed: hT[bt][d][m] (f16).
// grid: (BB*TT, NN/128), block 256. Wave w: m-tile w, all 8 d-tiles.
// ---------------------------------------------------------------------------
template <bool A_F32>
__global__ __launch_bounds__(256)
void akgcn_gemm1(const void* __restrict__ Ain,
                 const _Float16* __restrict__ thT,   // [CC][CC] per-layer
                 _Float16* __restrict__ hT) {
    constexpr int LDW = CC + 8;                 // 136-half padded row stride
    __shared__ _Float16 sTh[CC * LDW];          // 34,816 B

    const int bt   = blockIdx.x;
    const int m0   = blockIdx.y * 128;
    const int tid  = threadIdx.x;
    const int wave = tid >> 5;
    const int lane = tid & 31;
    const int ln   = lane & 15;
    const int abase = (lane < 16) ? 0 : 8;      // A operand K base
    const int bbase = (lane < 16) ? 0 : 16;     // unused; B also uses abase
    const int hoff  = abase;

    // Stage theta_T -> LDS: 128x128 halfs = 2048 8-half chunks, 8 per thread
    #pragma unroll
    for (int q = 0; q < 8; ++q) {
        const int idx = tid + 256 * q;
        const int r = idx >> 4;
        const int c = (idx & 15) * 8;
        *(half8*)&sTh[r * LDW + c] = *(const half8*)(thT + (size_t)r * CC + c);
    }
    __syncthreads();

    const int mA = m0 + 16 * wave + ln;
    const size_t arow = ((size_t)bt * NN + mA) * CC;

    float8 acc[8];
    const float8 zf = {0.f, 0.f, 0.f, 0.f, 0.f, 0.f, 0.f, 0.f};
    #pragma unroll
    for (int j = 0; j < 8; ++j) acc[j] = zf;

    half16 a_cur;
    {
        if constexpr (A_F32) {
            const float* Ap = ((const float*)Ain) + arow + abase;
            a_cur = h16_combine(cvt_f32_to_h8(Ap), cvt_f32_to_h8(Ap + 16));
        } else {
            const _Float16* Ap = ((const _Float16*)Ain) + arow + abase;
            a_cur = h16_combine(load_h8(Ap), load_h8(Ap + 16));
        }
    }

    #pragma unroll
    for (int kk = 0; kk < CC; kk += 32) {
        // prefetch next A fragment (global: long latency, issue first)
        half16 a_next = a_cur;
        if (kk + 32 < CC) {
            if constexpr (A_F32) {
                const float* Ap = ((const float*)Ain) + arow + kk + 32 + abase;
                a_next = h16_combine(cvt_f32_to_h8(Ap), cvt_f32_to_h8(Ap + 16));
            } else {
                const _Float16* Ap =
                    ((const _Float16*)Ain) + arow + kk + 32 + abase;
                a_next = h16_combine(load_h8(Ap), load_h8(Ap + 16));
            }
        }
        // load ALL 8 B tiles into distinct registers (ds clause)
        half16 bm[8];
        #pragma unroll
        for (int j = 0; j < 8; ++j) {
            const _Float16* Bp = &sTh[(size_t)(16 * j + ln) * LDW + kk + abase];
            bm[j] = h16_combine(load_h8(Bp), load_h8(Bp + 16));
        }
        // WMMA group: wait per-tile, staggered
        #pragma unroll
        for (int j = 0; j < 8; ++j) {
            acc[j] = __builtin_amdgcn_wmma_f32_16x16x32_f16(
                false, a_cur, false, bm[j], (short)0, acc[j], false, false);
        }
        a_cur = a_next;
    }

    // store transposed: hT[bt][d][m]; lane's 8 rows contiguous in m
    #pragma unroll
    for (int j = 0; j < 8; ++j) {
        const int d = 16 * j + ln;
        half8 hv;
        #pragma unroll
        for (int r = 0; r < 8; ++r) hv[r] = (_Float16)acc[j][r];
        _Float16* dst = hT + ((size_t)bt * CC + d) * NN + m0 + 16 * wave + hoff;
        *(half8*)dst = hv;
    }
}

// ---------------------------------------------------------------------------
// GEMM2: out[n,d] = sum_m V[n,m] * h[m,d]  as  out_T[d,n] = hT . V^T
//   A = hT[bt][d][m] (global, register double-buffered)
//   B = V rows (shared by all 8 waves) -> LDS, double-buffered 64-wide K slabs
//   All 4 B tiles of a k-step loaded into distinct regs before the WMMA group.
// Epilogue: ReLU -> hid (f16)  or  final: relu(x) + sigmoid(acc) -> f32 out
// grid: (BB*TT, NN/64), block 256. Wave w: d-tile w, n-tiles j=0..3.
// ---------------------------------------------------------------------------
template <bool FINAL>
__global__ __launch_bounds__(256)
void akgcn_gemm2(const _Float16* __restrict__ hT,
                 const _Float16* __restrict__ V,
                 const float* __restrict__ x,
                 _Float16* __restrict__ hid_out,
                 float* __restrict__ final_out) {
    constexpr int LDW = 72;                     // padded row stride (halfs)
    __shared__ _Float16 sB[2][64 * LDW];        // 18,432 B

    const int bt   = blockIdx.x;
    const int b    = bt >> 4;                   // / TT
    const int n0   = blockIdx.y * 64;
    const int tid  = threadIdx.x;
    const int wave = tid >> 5;
    const int lane = tid & 31;
    const int ln   = lane & 15;
    const int abase = (lane < 16) ? 0 : 8;
    const int bbase = (lane < 16) ? 0 : 16;
    const int hoff  = abase;

    // staging map: 64 rows x 64 halfs per slab = 512 8-half chunks, 2/thread
    const int sr = tid >> 3;                    // 0..31
    const int sc = (tid & 7) * 8;               // 0..56
    const _Float16* Vb = V + ((size_t)b * NN + n0) * NN;
    const _Float16* g0 = Vb + (size_t)sr * NN + sc;
    const _Float16* g1 = Vb + (size_t)(sr + 32) * NN + sc;

    const int dA = 16 * wave + ln;
    const _Float16* Arow = hT + ((size_t)bt * CC + dA) * NN;

    float8 acc[4];
    const float8 zf = {0.f, 0.f, 0.f, 0.f, 0.f, 0.f, 0.f, 0.f};
    #pragma unroll
    for (int j = 0; j < 4; ++j) acc[j] = zf;

    // ---- prologue: slab 0 -> LDS[0], first A fragment -> regs ----
    half8 p0 = *(const half8*)(g0);
    half8 p1 = *(const half8*)(g1);
    *(half8*)&sB[0][sr * LDW + sc]        = p0;
    *(half8*)&sB[0][(sr + 32) * LDW + sc] = p1;

    half16 a_cur;
    {
        const _Float16* Ap = Arow + abase;
        a_cur = h16_combine(load_h8(Ap), load_h8(Ap + 16));
    }

    constexpr int NSTAGE = NN / 64;             // 8
    for (int i = 0; i < NSTAGE; ++i) {
        const int s = i & 1;
        if (i + 1 < NSTAGE) {                   // prefetch next slab -> regs
            p0 = *(const half8*)(g0 + 64 * (i + 1));
            p1 = *(const half8*)(g1 + 64 * (i + 1));
        }
        __syncthreads();                        // LDS[s] stores visible

        #pragma unroll
        for (int sub = 0; sub < 2; ++sub) {
            // prefetch next A fragment (wraps harmlessly at the very end)
            int nk = 64 * i + 32 * sub + 32;
            if (nk >= NN) nk = 0;
            const _Float16* Ap = Arow + nk + abase;
            half16 a_next = h16_combine(load_h8(Ap), load_h8(Ap + 16));

            const int colb = 32 * sub + abase;
            // load ALL 4 B tiles into distinct registers (ds clause)
            half16 bm[4];
            #pragma unroll
            for (int j = 0; j < 4; ++j) {
                const _Float16* Bp =
                    &sB[s][(size_t)(16 * j + ln) * LDW + colb];
                bm[j] = h16_combine(load_h8(Bp), load_h8(Bp + 16));
            }
            #pragma unroll
            for (int j = 0; j < 4; ++j) {
                acc[j] = __builtin_amdgcn_wmma_f32_16x16x32_f16(
                    false, a_cur, false, bm[j], (short)0, acc[j], false, false);
            }
            a_cur = a_next;
        }

        if (i + 1 < NSTAGE) {
            __syncthreads();                    // all waves done reading sB[s^1]
            *(half8*)&sB[s ^ 1][sr * LDW + sc]        = p0;
            *(half8*)&sB[s ^ 1][(sr + 32) * LDW + sc] = p1;
        }
    }

    // D tile (rows d, cols n) -> row-major out[n][d]: contiguous in d per lane
    #pragma unroll
    for (int j = 0; j < 4; ++j) {
        const int n = n0 + 16 * j + ln;
        const size_t base = ((size_t)bt * NN + n) * CC + 16 * wave + hoff;
        if constexpr (FINAL) {
            float8 xv = *(const float8*)(x + base);
            float8 o;
            #pragma unroll
            for (int r = 0; r < 8; ++r) {
                const float sg = 1.0f / (1.0f + expf(-acc[j][r]));
                o[r] = fmaxf(xv[r], 0.0f) + sg;
            }
            *(float8*)(final_out + base) = o;
        } else {
            half8 hv;
            #pragma unroll
            for (int r = 0; r < 8; ++r)
                hv[r] = (_Float16)fmaxf(acc[j][r], 0.0f);
            *(half8*)(hid_out + base) = hv;
        }
    }
}

// ---------------------------------------------------------------------------
extern "C" void kernel_launch(void* const* d_in, const int* in_sizes, int n_in,
                              void* d_out, int out_size, void* d_ws, size_t ws_size,
                              hipStream_t stream) {
    (void)in_sizes; (void)n_in; (void)out_size; (void)ws_size;

    const float* x       = (const float*)d_in[0];   // [B,T,N,C]
    const float* adj     = (const float*)d_in[1];   // [B,N,N]
    const float* lambdas = (const float*)d_in[2];   // [K]
    const float* thetas  = (const float*)d_in[3];   // [K,C,C]
    float*       out     = (float*)d_out;           // [B,T,N,C]

    _Float16* thT = (_Float16*)d_ws;                         // K*C*C
    _Float16* V   = thT + (size_t)KK * CC * CC;              // B*N*N
    _Float16* hT  = V   + (size_t)BB * NN * NN;              // B*T*C*N
    _Float16* hid = hT  + (size_t)BB * TT * CC * NN;         // B*T*N*C
    // total ~151 MB of workspace

    const dim3 blk(256);
    const dim3 grid_g1(BB * TT, NN / 128);
    const dim3 grid_g2(BB * TT, NN / 64);

    akgcn_theta_prep<<<dim3(CC, KK), dim3(CC), 0, stream>>>(thetas, thT);

    // ---- layer 0 ----
    akgcn_compute_v<<<BB * NN, blk, 0, stream>>>(adj, lambdas, 0, V);
    akgcn_gemm1<true><<<grid_g1, blk, 0, stream>>>((const void*)x, thT, hT);
    akgcn_gemm2<false><<<grid_g2, blk, 0, stream>>>(hT, V, nullptr, hid, nullptr);

    // ---- layer 1 (final) ----
    akgcn_compute_v<<<BB * NN, blk, 0, stream>>>(adj, lambdas, 1, V);
    akgcn_gemm1<false><<<grid_g1, blk, 0, stream>>>((const void*)hid,
                                                    thT + (size_t)CC * CC, hT);
    akgcn_gemm2<true><<<grid_g2, blk, 0, stream>>>(hT, V, x, nullptr, out);
}
